// GAT_AT_9552007266594
// MI455X (gfx1250) — matrix operand order repaired and checked by reading, not compiled, not run on hardware
//
#include <hip/hip_runtime.h>
#include <hip/hip_bf16.h>

typedef __attribute__((ext_vector_type(16))) _Float16 v16h;
typedef __attribute__((ext_vector_type(8)))  float    v8f;
typedef __attribute__((ext_vector_type(2)))  _Float16 h2v;

#define NEG_SLOPE 0.2f

__device__ __forceinline__ void atomicMaxF(float* addr, float v) {
    // correct float max via monotone integer encodings (init must be finite)
    if (v >= 0.0f) atomicMax((int*)addr, __float_as_int(v));
    else           atomicMin((unsigned int*)addr, __float_as_uint(v));
}

// ---------------------------------------------------------------------------
// WMMA GEMM: out[M,NOUT] = x[M,K] @ W[K,NOUT]   (f16 multiply, f32 accumulate)
// One wave owns a 16-row strip and ALL NOUT output columns (NT=NOUT/16
// accumulators), so each strip of x is loaded exactly once. W is staged
// transposed in LDS as f16. A/B lane packing per CDNA5 ISA 7.12.2.
// ---------------------------------------------------------------------------
template<int K, int NOUT>
__global__ void __launch_bounds__(256)
gemm_wmma_f16(const float* __restrict__ x, const float* __restrict__ W,
              float* __restrict__ out, int M, int numRowTiles) {
    constexpr int NT = NOUT / 16;
    __shared__ _Float16 Wt[NOUT * K];          // [n][k] transposed
    const int t = threadIdx.x;
    for (int i = t; i < NOUT * K; i += blockDim.x) {
        const int n = i / K, k = i % K;
        Wt[i] = (_Float16)W[k * NOUT + n];
    }
    __syncthreads();

    const int lane = t & 31;
    const int wave = t >> 5;
    const int tile = blockIdx.x * (blockDim.x >> 5) + wave;   // row tile
    if (tile >= numRowTiles) return;           // wave-uniform: EXEC stays all-1s

    const int row0 = tile * 16;
    int m = row0 + (lane & 15);
    if (m >= M) m = M - 1;                     // clamp; stores guarded below
    const float* xrow = x + (size_t)m * K;
    const int ncol = lane & 15;

    const int ahi = (lane >> 4) << 3;          // A: lanes 16-31 handle K+8
    const int bhi = (lane >> 4) << 4;          // B: lanes 16-31 handle K+8 (same order as A)

    v8f c[NT] = {};
    #pragma unroll
    for (int kb = 0; kb < K; kb += 32) {
        v16h a;
        #pragma unroll
        for (int v = 0; v < 4; ++v) {          // compiler fuses into b128 loads
            const float2 f0 = *reinterpret_cast<const float2*>(xrow + kb + ahi + 2 * v);
            const float2 f1 = *reinterpret_cast<const float2*>(xrow + kb + 16 + ahi + 2 * v);
            a[2 * v]     = (_Float16)f0.x;  a[2 * v + 1] = (_Float16)f0.y;
            a[8 + 2 * v] = (_Float16)f1.x;  a[9 + 2 * v] = (_Float16)f1.y;
        }
        #pragma unroll
        for (int nt = 0; nt < NT; ++nt) {      // independent accumulators -> ILP
            const _Float16* wcol = Wt + (size_t)(nt * 16 + ncol) * K;
            v16h b;
            #pragma unroll
            for (int v = 0; v < 8; ++v) {
                const h2v w = *reinterpret_cast<const h2v*>(wcol + kb + ahi + 2 * v + ((v >> 2) << 3));
                b[2 * v] = w.x;  b[2 * v + 1] = w.y;
            }
            c[nt] = __builtin_amdgcn_wmma_f32_16x16x32_f16(false, a, false, b,
                                                           (short)0, c[nt], false, false);
        }
    }

    const int rbase = row0 + ((lane >> 4) << 3);   // D: VGPR r -> row r (+8 hi lanes)
    #pragma unroll
    for (int nt = 0; nt < NT; ++nt) {
        #pragma unroll
        for (int r = 0; r < 8; ++r) {
            const int row = rbase + r;
            if (row < M) out[(size_t)row * NOUT + nt * 16 + ncol] = c[nt][r];
        }
    }
}

// ---------------------------------------------------------------------------
// Per-node attention score dots + init of max / den / output accumulator.
// One thread per (node, head).
// ---------------------------------------------------------------------------
template<int H, int C>
__global__ void __launch_bounds__(256)
scores_init(const float* __restrict__ h, const float* __restrict__ a_src,
            const float* __restrict__ a_dst,
            float* __restrict__ sc_s, float* __restrict__ sc_d,
            float* __restrict__ mmax, float* __restrict__ den,
            float* __restrict__ outacc, int Nn) {
    const int tid = blockIdx.x * blockDim.x + threadIdx.x;
    if (tid >= Nn * H) return;
    const int n = tid / H, hh = tid % H;
    const float* hr = h + (size_t)n * (H * C) + hh * C;
    float ss = 0.f, sd = 0.f;
    #pragma unroll
    for (int cix = 0; cix < C; ++cix) {
        const float v = hr[cix];
        ss += v * a_src[hh * C + cix];
        sd += v * a_dst[hh * C + cix];
    }
    sc_s[tid] = ss;  sc_d[tid] = sd;
    mmax[tid] = -1e30f;  den[tid] = 0.f;
    float* oa = outacc + (size_t)n * (H * C) + hh * C;
    #pragma unroll
    for (int cix = 0; cix < C; ++cix) oa[cix] = 0.f;
}

// ---------------------------------------------------------------------------
// Edge pass 1: raw leaky-relu score -> escore buffer, atomic segment-max.
// ---------------------------------------------------------------------------
template<int H>
__global__ void __launch_bounds__(256)
edge_score_max(const int* __restrict__ src, const int* __restrict__ dst,
               const float* __restrict__ sc_s, const float* __restrict__ sc_d,
               float* __restrict__ mmax, float* __restrict__ escore,
               int E, int Etot) {
    const int tid = blockIdx.x * blockDim.x + threadIdx.x;
    if (tid >= Etot * H) return;
    const int e = tid / H, hh = tid % H;
    int s, d;
    if (e < E) { s = src[e]; d = dst[e]; } else { s = d = e - E; }  // self-loops
    float v = sc_s[s * H + hh] + sc_d[d * H + hh];
    v = v > 0.f ? v : NEG_SLOPE * v;
    escore[tid] = v;
    atomicMaxF(&mmax[d * H + hh], v);
}

// ---------------------------------------------------------------------------
// Edge pass 2: ex = exp(score - max[d]); accumulate denominator.
// ---------------------------------------------------------------------------
template<int H>
__global__ void __launch_bounds__(256)
edge_exp_sum(const int* __restrict__ dst, float* __restrict__ escore,
             const float* __restrict__ mmax, float* __restrict__ den,
             int E, int Etot) {
    const int tid = blockIdx.x * blockDim.x + threadIdx.x;
    if (tid >= Etot * H) return;
    const int e = tid / H, hh = tid % H;
    const int d = (e < E) ? dst[e] : (e - E);
    const float ex = __expf(escore[tid] - mmax[d * H + hh]);
    escore[tid] = ex;
    atomicAdd(&den[d * H + hh], ex);
}

// ---------------------------------------------------------------------------
// Edge pass 3: normalize -> final alpha (this is an output of the reference).
// ---------------------------------------------------------------------------
template<int H>
__global__ void __launch_bounds__(256)
edge_norm(const int* __restrict__ dst, float* __restrict__ alpha,
          const float* __restrict__ den, int E, int Etot) {
    const int tid = blockIdx.x * blockDim.x + threadIdx.x;
    if (tid >= Etot * H) return;
    const int e = tid / H, hh = tid % H;
    const int d = (e < E) ? dst[e] : (e - E);
    alpha[tid] = alpha[tid] / (den[d * H + hh] + 1e-16f);
}

// ---------------------------------------------------------------------------
// Edge pass 4: out[d] += h[s] * alpha  (scatter-add; accumulators fit in L2).
// CHUNKS threads per (edge, head), each covering C/CHUNKS channels.
// ---------------------------------------------------------------------------
template<int H, int C, int CHUNKS>
__global__ void __launch_bounds__(256)
edge_aggregate(const int* __restrict__ src, const int* __restrict__ dst,
               const float* __restrict__ alpha, const float* __restrict__ hfeat,
               float* __restrict__ outacc, int E, int Etot) {
    constexpr int CPT = C / CHUNKS;
    const int tid = blockIdx.x * blockDim.x + threadIdx.x;
    if (tid >= Etot * H * CHUNKS) return;
    const int e   = tid / (H * CHUNKS);
    const int rem = tid % (H * CHUNKS);
    const int hh  = rem / CHUNKS;
    const int ck  = rem % CHUNKS;
    int s, d;
    if (e < E) { s = src[e]; d = dst[e]; } else { s = d = e - E; }
    const float* hrow = hfeat + (size_t)s * (H * C) + hh * C + ck * CPT;
    __builtin_prefetch(hrow, 0, 0);            // global_prefetch_b8 on gathered row
    const float al = alpha[e * H + hh];
    float* orow = outacc + (size_t)d * (H * C) + hh * C + ck * CPT;
    #pragma unroll
    for (int i = 0; i < CPT; ++i) atomicAdd(&orow[i], hrow[i] * al);
}

// ---------------------------------------------------------------------------
// x1 = elu(out1 + b1), in place.
// ---------------------------------------------------------------------------
template<int F>
__global__ void __launch_bounds__(256)
elu_bias(float* __restrict__ xio, const float* __restrict__ b, int total) {
    const int i = blockIdx.x * blockDim.x + threadIdx.x;
    if (i >= total) return;
    const float v = xio[i] + b[i % F];
    xio[i] = v > 0.f ? v : (__expf(v) - 1.f);
}

// ---------------------------------------------------------------------------
// logits = out2 + b2 ; logp = log_softmax(logits). One thread per node.
// ---------------------------------------------------------------------------
__global__ void __launch_bounds__(256)
final_logsoftmax(const float* __restrict__ out2, const float* __restrict__ b2,
                 float* __restrict__ logp, float* __restrict__ logits, int Nn) {
    const int n = blockIdx.x * blockDim.x + threadIdx.x;
    if (n >= Nn) return;
    float vals[32];
    float mx = -1e30f;
    #pragma unroll
    for (int c = 0; c < 32; ++c) {
        const float v = out2[(size_t)n * 32 + c] + b2[c];
        vals[c] = v;
        mx = v > mx ? v : mx;
    }
    float ssum = 0.f;
    #pragma unroll
    for (int c = 0; c < 32; ++c) ssum += __expf(vals[c] - mx);
    const float lse = mx + __logf(ssum);
    #pragma unroll
    for (int c = 0; c < 32; ++c) {
        logits[(size_t)n * 32 + c] = vals[c];
        logp[(size_t)n * 32 + c]   = vals[c] - lse;
    }
}

// ---------------------------------------------------------------------------
extern "C" void kernel_launch(void* const* d_in, const int* in_sizes, int n_in,
                              void* d_out, int out_size, void* d_ws, size_t ws_size,
                              hipStream_t stream) {
    const float* x      = (const float*)d_in[0];
    const int*   src    = (const int*)  d_in[1];
    const int*   dst    = (const int*)  d_in[2];
    const float* W1     = (const float*)d_in[3];
    const float* a_src1 = (const float*)d_in[4];
    const float* a_dst1 = (const float*)d_in[5];
    const float* b1     = (const float*)d_in[6];
    const float* W2     = (const float*)d_in[7];
    const float* a_src2 = (const float*)d_in[8];
    const float* a_dst2 = (const float*)d_in[9];
    const float* b2     = (const float*)d_in[10];

    const int Nn   = in_sizes[0] / 256;     // nodes
    const int E    = in_sizes[1];           // random edges
    const int Etot = E + Nn;                // + self-loops

    // ---- outputs, packed in reference return order
    float* out    = (float*)d_out;
    float* logp   = out;
    float* alpha1 = out + (size_t)Nn * 32;
    float* alpha2 = alpha1 + (size_t)Etot * 8;
    float* logits = alpha2 + (size_t)Etot;

    // ---- workspace layout
    float* ws   = (float*)d_ws;
    float* h1   = ws;  ws += (size_t)Nn * 64;
    float* sc1s = ws;  ws += (size_t)Nn * 8;
    float* sc1d = ws;  ws += (size_t)Nn * 8;
    float* m1   = ws;  ws += (size_t)Nn * 8;
    float* den1 = ws;  ws += (size_t)Nn * 8;
    float* out1 = ws;  ws += (size_t)Nn * 64;   // becomes x1 after elu
    float* h2   = ws;  ws += (size_t)Nn * 32;
    float* sc2s = ws;  ws += (size_t)Nn;
    float* sc2d = ws;  ws += (size_t)Nn;
    float* m2   = ws;  ws += (size_t)Nn;
    float* den2 = ws;  ws += (size_t)Nn;
    float* out2 = ws;  ws += (size_t)Nn * 32;

    const int TB = 256;
    auto nblk = [](long long threads, int tb) { return (int)((threads + tb - 1) / tb); };

    const int rowTiles = (Nn + 15) / 16;

    // ===================== Layer 1 (H=8, C=8) =====================
    gemm_wmma_f16<256, 64><<<nblk((long long)rowTiles * 32, TB), TB, 0, stream>>>(
        x, W1, h1, Nn, rowTiles);
    scores_init<8, 8><<<nblk((long long)Nn * 8, TB), TB, 0, stream>>>(
        h1, a_src1, a_dst1, sc1s, sc1d, m1, den1, out1, Nn);
    edge_score_max<8><<<nblk((long long)Etot * 8, TB), TB, 0, stream>>>(
        src, dst, sc1s, sc1d, m1, alpha1, E, Etot);
    edge_exp_sum<8><<<nblk((long long)Etot * 8, TB), TB, 0, stream>>>(
        dst, alpha1, m1, den1, E, Etot);
    edge_norm<8><<<nblk((long long)Etot * 8, TB), TB, 0, stream>>>(
        dst, alpha1, den1, E, Etot);
    edge_aggregate<8, 8, 1><<<nblk((long long)Etot * 8, TB), TB, 0, stream>>>(
        src, dst, alpha1, h1, out1, E, Etot);
    elu_bias<64><<<nblk((long long)Nn * 64, TB), TB, 0, stream>>>(out1, b1, Nn * 64);

    // ===================== Layer 2 (H=1, C=32) =====================
    gemm_wmma_f16<64, 32><<<nblk((long long)rowTiles * 32, TB), TB, 0, stream>>>(
        out1, W2, h2, Nn, rowTiles);
    scores_init<1, 32><<<nblk((long long)Nn, TB), TB, 0, stream>>>(
        h2, a_src2, a_dst2, sc2s, sc2d, m2, den2, out2, Nn);
    edge_score_max<1><<<nblk((long long)Etot, TB), TB, 0, stream>>>(
        src, dst, sc2s, sc2d, m2, alpha2, E, Etot);
    edge_exp_sum<1><<<nblk((long long)Etot, TB), TB, 0, stream>>>(
        dst, alpha2, m2, den2, E, Etot);
    edge_norm<1><<<nblk((long long)Etot, TB), TB, 0, stream>>>(
        dst, alpha2, den2, E, Etot);
    edge_aggregate<1, 32, 4><<<nblk((long long)Etot * 4, TB), TB, 0, stream>>>(
        src, dst, alpha2, h2, out2, E, Etot);

    final_logsoftmax<<<nblk((long long)Nn, TB), TB, 0, stream>>>(
        out2, b2, logp, logits, Nn);
}
